// Mamba_Traj_18253611008298
// MI455X (gfx1250) — matrix operand order repaired and checked by reading, not compiled, not run on hardware
//
#include <hip/hip_runtime.h>
#include <hip/hip_bf16.h>
#include <math.h>

typedef __attribute__((ext_vector_type(16))) _Float16 v16h;
typedef __attribute__((ext_vector_type(8)))  _Float16 v8h;
typedef __attribute__((ext_vector_type(4)))  _Float16 v4h;
typedef __attribute__((ext_vector_type(8)))  float    v8f;
typedef __attribute__((ext_vector_type(4)))  float    v4f;
typedef __attribute__((ext_vector_type(2)))  float    v2f;

#define BN      8
#define LS      128
#define LP      384          // 3*LS tokens per batch after interleave
#define TOK     3072         // BN*LP
#define EDIM    256
#define DMODEL  256
#define NLAYER  4
#define DINNER  512
#define DSTATE  16
#define RKDT    16
#define DCONV   4
#define PROJW   48           // RKDT + 2*DSTATE

__device__ __forceinline__ float siluf(float x) { return x / (1.0f + __expf(-x)); }

// ---------------------------------------------------------------------------
// Embedding + timestep gather + (g,s,a) interleave -> u (TOK, EDIM)
// ---------------------------------------------------------------------------
__global__ __launch_bounds__(256) void embed_kernel(
    const float* __restrict__ states, const float* __restrict__ actions,
    const float* __restrict__ goal,   const int* __restrict__ tsteps,
    const float* __restrict__ teW,
    const float* __restrict__ seW, const float* __restrict__ seb,
    const float* __restrict__ geW, const float* __restrict__ geb,
    const float* __restrict__ aeW, const float* __restrict__ aeb,
    float* __restrict__ u)
{
  int idx = blockIdx.x * 256 + threadIdx.x;   // over TOK*EDIM
  int e   = idx & (EDIM - 1);
  int tt  = idx >> 8;                         // token index 0..TOK-1
  int slot = tt % 3;                          // 0=goal, 1=state, 2=action
  int bl   = tt / 3;                          // b*LS + l
  float acc;
  if (slot == 0) {
    acc = geb[e];
    #pragma unroll
    for (int k = 0; k < 6; ++k) acc += goal[bl * 6 + k] * geW[k * EDIM + e];
  } else if (slot == 1) {
    acc = seb[e];
    #pragma unroll
    for (int k = 0; k < 6; ++k) acc += states[bl * 6 + k] * seW[k * EDIM + e];
  } else {
    acc = aeb[e];
    #pragma unroll
    for (int k = 0; k < 3; ++k) acc += actions[bl * 3 + k] * aeW[k * EDIM + e];
  }
  acc += teW[(size_t)tsteps[bl] * EDIM + e];
  u[idx] = acc;
}

// ---------------------------------------------------------------------------
// WMMA GEMM: Out = A(MxK) * Bw(KxN) [+ bias] [+ Cin]   (f32 in memory, f16
// operands staged through LDS, f32 accumulation in the WMMA C/D operand)
// Block = 256 threads = 8 waves; tile 128(M) x 16(N), K in chunks of 32.
// Requires M%128==0, N%16==0 (N even), K%32==0 (true at all call sites).
// Staging is fixed-trip + vectorized: b128 global loads, b64 LDS stores.
// ---------------------------------------------------------------------------
#define TM 128
#define TN 16
#define TK 32
#define LDA 40   // halves; 80B row stride -> conflict-free b128 LDS reads
#define LDB 40

__global__ __launch_bounds__(256) void gemm_wmma_kernel(
    const float* __restrict__ A, const float* __restrict__ Bw,
    const float* __restrict__ bias, const float* __restrict__ Cin,
    float* __restrict__ Out, int M, int N, int K)
{
  __shared__ _Float16 As[TM * LDA];
  __shared__ _Float16 Bs[TN * LDB];
  const int tid  = threadIdx.x;
  const int wave = tid >> 5;
  const int lane = tid & 31;
  const int m0 = blockIdx.x * TM;
  const int n0 = blockIdx.y * TN;

  // C/D layout: element r <-> row = r + (lane>=16 ? 8:0), col = lane&15
  const int crow = m0 + wave * 16 + ((lane & 16) ? 8 : 0);
  const int ccol = n0 + (lane & 15);

  v8f c;
  #pragma unroll
  for (int r = 0; r < 8; ++r)
    c[r] = Cin ? Cin[(size_t)(crow + r) * N + ccol] : 0.0f;
  if (bias) {
    float bv = bias[ccol];
    #pragma unroll
    for (int r = 0; r < 8; ++r) c[r] += bv;
  }

  // Precomputed staging coordinates (constant across K chunks)
  const int a_r  = tid >> 1;                 // A: thread -> row 0..127
  const int a_c4 = (tid & 1) << 4;           //    two 4-float groups per row pair? no:
  // (a_r, a_c4) unused; per-iteration vid form used below for clarity.
  (void)a_r; (void)a_c4;
  const int b_r  = tid >> 3;                 // B: row 0..31
  const int b_cc = (tid & 7) << 1;           //    col 0,2,...,14

  const int arow  = lane & 15;
  const int akoff = (lane & 16) ? 8 : 0;     // A: K-half select per lane group
  const int bcol  = lane & 15;
  const int bkoff = (lane & 16) ? 8 : 0;     // B: K-half select per lane group

  for (int k0 = 0; k0 < K; k0 += TK) {
    __syncthreads();
    // Stage A tile (128x32 f32 -> f16): 1024 float4s, 4 per thread.
    #pragma unroll
    for (int it = 0; it < 4; ++it) {
      int vid = tid + it * 256;              // 0..1023
      int r   = vid >> 3;                    // 8 float4 per 32-wide row
      int c4  = (vid & 7) << 2;
      v4f v = *(const v4f*)&A[(size_t)(m0 + r) * K + (k0 + c4)];
      v4h h;
      #pragma unroll
      for (int j = 0; j < 4; ++j) h[j] = (_Float16)v[j];
      *(v4h*)&As[r * LDA + c4] = h;          // 8B aligned (LDA*2=80, c4*2 in {0..56})
    }
    // Stage B tile (32x16 f32) transposed into Bs[col][k]: 256 float2s.
    {
      v2f v = *(const v2f*)&Bw[(size_t)(k0 + b_r) * N + (n0 + b_cc)];
      Bs[(b_cc + 0) * LDB + b_r] = (_Float16)v[0];
      Bs[(b_cc + 1) * LDB + b_r] = (_Float16)v[1];
    }
    if (k0 + TK < K)  // global_prefetch_b8 of next A K-slab
      __builtin_prefetch(&A[(size_t)(m0 + (tid >> 1)) * K + (k0 + TK) + ((tid & 1) << 4)], 0, 1);
    __syncthreads();

    // A frag: elems 0..7 -> K=akoff+0..7 ; elems 8..15 -> K=16+akoff+0..7
    const _Float16* pa = &As[(wave * 16 + arow) * LDA];
    v8h a_lo = *(const v8h*)(pa + akoff);
    v8h a_hi = *(const v8h*)(pa + 16 + akoff);
    // B frag: elems 0..7 -> K=bkoff+0..7 ; elems 8..15 -> K=16+bkoff+0..7
    const _Float16* pb = &Bs[bcol * LDB + bkoff];
    v8h b_lo = *(const v8h*)(pb);
    v8h b_hi = *(const v8h*)(pb + 16);
    v16h af, bf;
    #pragma unroll
    for (int i = 0; i < 8; ++i) {
      af[i] = a_lo[i]; af[8 + i] = a_hi[i];
      bf[i] = b_lo[i]; bf[8 + i] = b_hi[i];
    }
    c = __builtin_amdgcn_wmma_f32_16x16x32_f16(false, af, false, bf,
                                               (short)0, c, false, false);
  }

  #pragma unroll
  for (int r = 0; r < 8; ++r)
    Out[(size_t)(crow + r) * N + ccol] = c[r];
}

// ---------------------------------------------------------------------------
// RMSNorm over D=256 per token (one block per token)
// ---------------------------------------------------------------------------
__global__ __launch_bounds__(256) void rmsnorm_kernel(
    const float* __restrict__ x, const float* __restrict__ w,
    float* __restrict__ out)
{
  int t = blockIdx.x;
  int i = threadIdx.x;
  __shared__ float red[256];
  float v = x[(size_t)t * DMODEL + i];
  red[i] = v * v;
  __syncthreads();
  #pragma unroll
  for (int s = 128; s > 0; s >>= 1) {
    if (i < s) red[i] += red[i + s];
    __syncthreads();
  }
  float scale = rsqrtf(red[0] * (1.0f / DMODEL) + 1e-5f);
  out[(size_t)t * DMODEL + i] = v * scale * w[i];
}

// ---------------------------------------------------------------------------
// Depthwise causal conv1d (kernel 4, left pad 3) + SiLU.  xc slice of xz.
// ---------------------------------------------------------------------------
__global__ __launch_bounds__(256) void conv_silu_kernel(
    const float* __restrict__ xz, const float* __restrict__ cw,
    const float* __restrict__ cb, float* __restrict__ xc)
{
  int idx = blockIdx.x * 256 + threadIdx.x;   // over TOK*DINNER
  int d = idx & (DINNER - 1);
  int t = idx >> 9;
  int l = t % LP;
  float acc = cb[d];
  #pragma unroll
  for (int k = 0; k < DCONV; ++k) {
    int ls = l + k - (DCONV - 1);             // w[3] hits current sample
    if (ls >= 0)
      acc += xz[(size_t)(t + ls - l) * (2 * DINNER) + d] * cw[d * DCONV + k];
  }
  xc[idx] = siluf(acc);
}

// ---------------------------------------------------------------------------
// dt = softplus(proj[:, :16] @ dt_W + dt_b)
// ---------------------------------------------------------------------------
__global__ __launch_bounds__(256) void dt_softplus_kernel(
    const float* __restrict__ proj, const float* __restrict__ dtW,
    const float* __restrict__ dtb,  float* __restrict__ dt)
{
  int idx = blockIdx.x * 256 + threadIdx.x;   // over TOK*DINNER
  int d = idx & (DINNER - 1);
  int t = idx >> 9;
  float acc = dtb[d];
  #pragma unroll
  for (int r = 0; r < RKDT; ++r)
    acc += proj[(size_t)t * PROJW + r] * dtW[r * DINNER + d];
  dt[idx] = (acc > 20.0f) ? acc : log1pf(__expf(acc));
}

// ---------------------------------------------------------------------------
// Selective scan. One block per batch element, 2 channels/thread, h & A in
// registers, B_t/C_t broadcast via LDS. Fuses +xc*Dp and *silu(z).
// ---------------------------------------------------------------------------
__global__ __launch_bounds__(256) void scan_kernel(
    const float* __restrict__ dt,  const float* __restrict__ proj,
    const float* __restrict__ xc,  const float* __restrict__ xz,
    const float* __restrict__ Alog, const float* __restrict__ Dp,
    float* __restrict__ y)
{
  const int b = blockIdx.x;
  const int tid = threadIdx.x;
  const int d0 = tid, d1 = tid + 256;
  float A0[DSTATE], A1[DSTATE], h0[DSTATE], h1[DSTATE];
  #pragma unroll
  for (int n = 0; n < DSTATE; ++n) {
    A0[n] = -__expf(Alog[d0 * DSTATE + n]);
    A1[n] = -__expf(Alog[d1 * DSTATE + n]);
    h0[n] = 0.0f; h1[n] = 0.0f;
  }
  const float Dv0 = Dp[d0], Dv1 = Dp[d1];
  __shared__ float sB[DSTATE], sC[DSTATE];

  for (int l = 0; l < LP; ++l) {
    const size_t t = (size_t)b * LP + l;
    __syncthreads();
    if (tid < DSTATE)          sB[tid] = proj[t * PROJW + RKDT + tid];
    else if (tid < 2 * DSTATE) sC[tid - DSTATE] = proj[t * PROJW + RKDT + DSTATE + (tid - DSTATE)];
    __syncthreads();
    const float dt0 = dt[t * DINNER + d0], dt1 = dt[t * DINNER + d1];
    const float x0  = xc[t * DINNER + d0], x1  = xc[t * DINNER + d1];
    const float bx0 = dt0 * x0, bx1 = dt1 * x1;
    float y0 = 0.0f, y1 = 0.0f;
    #pragma unroll
    for (int n = 0; n < DSTATE; ++n) {
      h0[n] = __expf(dt0 * A0[n]) * h0[n] + bx0 * sB[n];
      y0 += h0[n] * sC[n];
      h1[n] = __expf(dt1 * A1[n]) * h1[n] + bx1 * sB[n];
      y1 += h1[n] * sC[n];
    }
    y0 += x0 * Dv0;
    y1 += x1 * Dv1;
    const float z0 = xz[t * (2 * DINNER) + DINNER + d0];
    const float z1 = xz[t * (2 * DINNER) + DINNER + d1];
    y[t * DINNER + d0] = y0 * siluf(z0);
    y[t * DINNER + d1] = y1 * siluf(z1);
  }
}

// ---------------------------------------------------------------------------
// Heads: state_preds (a-slot tokens) then tanh(action_preds) (s-slot tokens)
// ---------------------------------------------------------------------------
__global__ __launch_bounds__(256) void head_kernel(
    const float* __restrict__ yf,
    const float* __restrict__ psW, const float* __restrict__ psb,
    const float* __restrict__ paW, const float* __restrict__ pab,
    float* __restrict__ out)
{
  int idx = blockIdx.x * 256 + threadIdx.x;   // over BN*LS*(6+3) = 9216
  const int NS = BN * LS * 6;
  if (idx < NS) {
    int j = idx % 6, bl = idx / 6;
    int b = bl / LS, l = bl % LS;
    const float* row = yf + (size_t)(b * LP + 3 * l + 2) * DMODEL;
    float acc = psb[j];
    for (int k = 0; k < DMODEL; ++k) acc += row[k] * psW[k * 6 + j];
    out[idx] = acc;
  } else {
    int i2 = idx - NS;
    int j = i2 % 3, bl = i2 / 3;
    int b = bl / LS, l = bl % LS;
    const float* row = yf + (size_t)(b * LP + 3 * l + 1) * DMODEL;
    float acc = pab[j];
    for (int k = 0; k < DMODEL; ++k) acc += row[k] * paW[k * 3 + j];
    out[idx] = tanhf(acc);
  }
}

// ---------------------------------------------------------------------------
extern "C" void kernel_launch(void* const* d_in, const int* in_sizes, int n_in,
                              void* d_out, int out_size, void* d_ws, size_t ws_size,
                              hipStream_t stream)
{
  (void)in_sizes; (void)n_in; (void)out_size; (void)ws_size;
  const float* states  = (const float*)d_in[0];
  const float* actions = (const float*)d_in[1];
  const float* goal    = (const float*)d_in[2];
  const int*   tsteps  = (const int*)  d_in[3];
  const float* teW   = (const float*)d_in[4];
  const float* seW   = (const float*)d_in[5];
  const float* seb   = (const float*)d_in[6];
  const float* geW   = (const float*)d_in[7];
  const float* geb   = (const float*)d_in[8];
  const float* aeW   = (const float*)d_in[9];
  const float* aeb   = (const float*)d_in[10];
  const float* bbW   = (const float*)d_in[11];
  const float* bbb   = (const float*)d_in[12];
  const float* normw = (const float*)d_in[13];
  const float* inW   = (const float*)d_in[14];
  const float* cw    = (const float*)d_in[15];
  const float* cb    = (const float*)d_in[16];
  const float* xpW   = (const float*)d_in[17];
  const float* dtW   = (const float*)d_in[18];
  const float* dtb   = (const float*)d_in[19];
  const float* Alog  = (const float*)d_in[20];
  const float* Dp    = (const float*)d_in[21];
  const float* outW  = (const float*)d_in[22];
  const float* fnw   = (const float*)d_in[23];
  const float* psW   = (const float*)d_in[24];
  const float* psb   = (const float*)d_in[25];
  const float* paW   = (const float*)d_in[26];
  const float* pab   = (const float*)d_in[27];

  float* ws = (float*)d_ws;
  float* u     = ws; ws += (size_t)TOK * EDIM;
  float* resid = ws; ws += (size_t)TOK * DMODEL;
  float* xn    = ws; ws += (size_t)TOK * DMODEL;
  float* xz    = ws; ws += (size_t)TOK * 2 * DINNER;
  float* xcb   = ws; ws += (size_t)TOK * DINNER;
  float* projb = ws; ws += (size_t)TOK * PROJW;
  float* dtv   = ws; ws += (size_t)TOK * DINNER;
  float* yb    = ws; ws += (size_t)TOK * DINNER;
  float* yf    = ws; ws += (size_t)TOK * DMODEL;

  embed_kernel<<<TOK * EDIM / 256, 256, 0, stream>>>(
      states, actions, goal, tsteps, teW, seW, seb, geW, geb, aeW, aeb, u);

  // residual = u @ bb_in_W + bb_in_b
  gemm_wmma_kernel<<<dim3(TOK / TM, DMODEL / TN), 256, 0, stream>>>(
      u, bbW, bbb, nullptr, resid, TOK, DMODEL, EDIM);

  for (int i = 0; i < NLAYER; ++i) {
    rmsnorm_kernel<<<TOK, 256, 0, stream>>>(resid, normw + (size_t)i * DMODEL, xn);
    // xz = xn @ in_proj_W
    gemm_wmma_kernel<<<dim3(TOK / TM, (2 * DINNER) / TN), 256, 0, stream>>>(
        xn, inW + (size_t)i * DMODEL * 2 * DINNER, nullptr, nullptr, xz,
        TOK, 2 * DINNER, DMODEL);
    conv_silu_kernel<<<TOK * DINNER / 256, 256, 0, stream>>>(
        xz, cw + (size_t)i * DINNER * DCONV, cb + (size_t)i * DINNER, xcb);
    // proj = xc @ xproj_W
    gemm_wmma_kernel<<<dim3(TOK / TM, PROJW / TN), 256, 0, stream>>>(
        xcb, xpW + (size_t)i * DINNER * PROJW, nullptr, nullptr, projb,
        TOK, PROJW, DINNER);
    dt_softplus_kernel<<<TOK * DINNER / 256, 256, 0, stream>>>(
        projb, dtW + (size_t)i * RKDT * DINNER, dtb + (size_t)i * DINNER, dtv);
    scan_kernel<<<BN, 256, 0, stream>>>(
        dtv, projb, xcb, xz, Alog + (size_t)i * DINNER * DSTATE,
        Dp + (size_t)i * DINNER, yb);
    // residual += y @ out_W   (residual fed through WMMA C operand)
    gemm_wmma_kernel<<<dim3(TOK / TM, DMODEL / TN), 256, 0, stream>>>(
        yb, outW + (size_t)i * DINNER * DMODEL, nullptr, resid, resid,
        TOK, DMODEL, DINNER);
  }

  rmsnorm_kernel<<<TOK, 256, 0, stream>>>(resid, fnw, yf);
  head_kernel<<<(BN * LS * 9) / 256, 256, 0, stream>>>(
      yf, psW, psb, paW, pab, (float*)d_out);
}
